// ImprovedHGTEncoder_41729902248517
// MI455X (gfx1250) — matrix operand order repaired and checked by reading, not compiled, not run on hardware
//
#include <hip/hip_runtime.h>
#include <math.h>

typedef _Float16 f16;
typedef __attribute__((ext_vector_type(16))) _Float16 v16h;
typedef __attribute__((ext_vector_type(8)))  float    v8f;

// =====================================================================
// Tiled f16 WMMA GEMM: C[M,N] = A[M,K] @ B[K,N] (+bias, optional relu)
// B is supplied PRE-TRANSPOSED: BT[N][ldb], ldb >= K (row = output col).
// Batched via blockIdx.z with element strides sAb/sBb/sCb.
// Block tile 128x128, K-step 32, 256 threads = 8 waves (wave32).
// Register-staged, double-buffered LDS pipeline.
// =====================================================================
__global__ __launch_bounds__(256) void hgt_gemm_f16(
    const f16* __restrict__ A, int lda, long long sAb,
    const f16* __restrict__ BT, int ldb, long long sBb,
    float* Cf, f16* Ch, int ldc, long long sCb,
    const float* __restrict__ bias,
    int M, int N, int K, int relu)
{
    __shared__ f16 As[2][128][40];   // [buf][row][k]   (+pad, 80B pitch)
    __shared__ f16 Bs[2][128][40];   // [buf][col][k]   (+pad)

    const int bz = blockIdx.z;
    A  += (size_t)bz * sAb;
    BT += (size_t)bz * sBb;
    const size_t coff = (size_t)bz * sCb;

    const int m0 = blockIdx.y * 128;
    const int n0 = blockIdx.x * 128;
    const int tid  = threadIdx.x;
    const int lane = tid & 31;
    const int wave = tid >> 5;
    const int wm = (wave & 3) * 32;   // wave's M offset inside tile (2 m-tiles)
    const int wn = (wave >> 2) * 64;  // wave's N offset inside tile (4 n-tiles)

    const bool inM = (m0 + 128 <= M);
    const bool inN = (n0 + 128 <= N);

    // staging coords: chunk c covers (row = c>>2, k-col = (c&3)*8), 16B each
    const int sr0 = tid >> 2;             // chunk 0 row (0..63)
    const int sr1 = sr0 + 64;             // chunk 1 row (64..127)
    const int sc  = (tid & 3) << 3;       // k column (0/8/16/24)

    // ISA 7.12.2 fragment addressing (wave32):
    // A: lane<16 -> row=lane,    K {0..7, 16..23}; lane>=16 -> row=lane-16, K {8..15, 24..31}
    // B: lane<16 -> col=lane,    K {0..7, 16..23}; lane>=16 -> col=lane-16, K {8..15, 24..31}
    const int khA   = (lane < 16) ? 0 : 8;
    const int kbB   = (lane < 16) ? 0 : 8;
    const int rowA0 = wm + (lane & 15);
    const int colB0 = wn + (lane & 15);

    v8f acc[2][4] = {};
    uint4 ra0, ra1, rb0, rb1;

    const int nk = K >> 5;

    // ---- prologue: load tile 0 into registers ----
    {
        const int k0 = 0;
        if (inM) {
            ra0 = *(const uint4*)(A + (size_t)(m0 + sr0) * lda + k0 + sc);
            ra1 = *(const uint4*)(A + (size_t)(m0 + sr1) * lda + k0 + sc);
        } else {
            ra0 = make_uint4(0u, 0u, 0u, 0u);
            ra1 = make_uint4(0u, 0u, 0u, 0u);
            if (m0 + sr0 < M) ra0 = *(const uint4*)(A + (size_t)(m0 + sr0) * lda + k0 + sc);
            if (m0 + sr1 < M) ra1 = *(const uint4*)(A + (size_t)(m0 + sr1) * lda + k0 + sc);
        }
        if (inN) {
            rb0 = *(const uint4*)(BT + (size_t)(n0 + sr0) * ldb + k0 + sc);
            rb1 = *(const uint4*)(BT + (size_t)(n0 + sr1) * ldb + k0 + sc);
        } else {
            rb0 = make_uint4(0u, 0u, 0u, 0u);
            rb1 = make_uint4(0u, 0u, 0u, 0u);
            if (n0 + sr0 < N) rb0 = *(const uint4*)(BT + (size_t)(n0 + sr0) * ldb + k0 + sc);
            if (n0 + sr1 < N) rb1 = *(const uint4*)(BT + (size_t)(n0 + sr1) * ldb + k0 + sc);
        }
    }

    int buf = 0;
    for (int t = 0; t < nk; ++t) {
        // ---- stage current tile regs -> LDS[buf] ----
        *(uint4*)(&As[buf][sr0][sc]) = ra0;
        *(uint4*)(&As[buf][sr1][sc]) = ra1;
        *(uint4*)(&Bs[buf][sr0][sc]) = rb0;
        *(uint4*)(&Bs[buf][sr1][sc]) = rb1;
        __syncthreads();

        // ---- issue global loads for next tile while computing this one ----
        if (t + 1 < nk) {
            const int k0 = (t + 1) << 5;
            if (inM) {
                ra0 = *(const uint4*)(A + (size_t)(m0 + sr0) * lda + k0 + sc);
                ra1 = *(const uint4*)(A + (size_t)(m0 + sr1) * lda + k0 + sc);
            } else {
                ra0 = make_uint4(0u, 0u, 0u, 0u);
                ra1 = make_uint4(0u, 0u, 0u, 0u);
                if (m0 + sr0 < M) ra0 = *(const uint4*)(A + (size_t)(m0 + sr0) * lda + k0 + sc);
                if (m0 + sr1 < M) ra1 = *(const uint4*)(A + (size_t)(m0 + sr1) * lda + k0 + sc);
            }
            if (inN) {
                rb0 = *(const uint4*)(BT + (size_t)(n0 + sr0) * ldb + k0 + sc);
                rb1 = *(const uint4*)(BT + (size_t)(n0 + sr1) * ldb + k0 + sc);
            } else {
                rb0 = make_uint4(0u, 0u, 0u, 0u);
                rb1 = make_uint4(0u, 0u, 0u, 0u);
                if (n0 + sr0 < N) rb0 = *(const uint4*)(BT + (size_t)(n0 + sr0) * ldb + k0 + sc);
                if (n0 + sr1 < N) rb1 = *(const uint4*)(BT + (size_t)(n0 + sr1) * ldb + k0 + sc);
            }
        }

        // ---- fragments + 8 WMMAs ----
        union Fr { uint4 q[2]; v16h v; };
        Fr a[2], b[4];
        #pragma unroll
        for (int mt = 0; mt < 2; ++mt) {
            const f16* p = &As[buf][rowA0 + mt * 16][khA];
            a[mt].q[0] = *(const uint4*)(p);        // K khA..khA+7
            a[mt].q[1] = *(const uint4*)(p + 16);   // K khA+16..khA+23
        }
        #pragma unroll
        for (int nt = 0; nt < 4; ++nt) {
            const f16* p = &Bs[buf][colB0 + nt * 16][kbB];
            b[nt].q[0] = *(const uint4*)(p);        // K kbB..kbB+7
            b[nt].q[1] = *(const uint4*)(p + 16);   // K kbB+16..kbB+23
        }
        #pragma unroll
        for (int mt = 0; mt < 2; ++mt)
            #pragma unroll
            for (int nt = 0; nt < 4; ++nt)
                acc[mt][nt] = __builtin_amdgcn_wmma_f32_16x16x32_f16(
                    false, a[mt].v, false, b[nt].v, (short)0, acc[mt][nt], false, false);
        buf ^= 1;
    }

    // ---- store: VGPR r -> (M = r + 8*(lane>=16), N = lane&15) per 16x16 tile ----
    const int rb2 = (lane < 16) ? 0 : 8;
    const int cl  = lane & 15;
    if (inM && inN) {
        #pragma unroll
        for (int mt = 0; mt < 2; ++mt)
            #pragma unroll
            for (int nt = 0; nt < 4; ++nt) {
                const int gn = n0 + wn + nt * 16 + cl;
                const float bv = bias ? bias[gn] : 0.0f;
                #pragma unroll
                for (int r = 0; r < 8; ++r) {
                    const int gm = m0 + wm + mt * 16 + rb2 + r;
                    float v = acc[mt][nt][r] + bv;
                    if (relu) v = fmaxf(v, 0.0f);
                    size_t o = coff + (size_t)gm * ldc + gn;
                    if (Cf) Cf[o] = v;
                    if (Ch) Ch[o] = (f16)v;
                }
            }
    } else {
        #pragma unroll
        for (int mt = 0; mt < 2; ++mt)
            #pragma unroll
            for (int nt = 0; nt < 4; ++nt) {
                const int gn = n0 + wn + nt * 16 + cl;
                const float bv = (bias && gn < N) ? bias[gn] : 0.0f;
                #pragma unroll
                for (int r = 0; r < 8; ++r) {
                    const int gm = m0 + wm + mt * 16 + rb2 + r;
                    if (gm < M && gn < N) {
                        float v = acc[mt][nt][r] + bv;
                        if (relu) v = fmaxf(v, 0.0f);
                        size_t o = coff + (size_t)gm * ldc + gn;
                        if (Cf) Cf[o] = v;
                        if (Ch) Ch[o] = (f16)v;
                    }
                }
            }
    }
}

// =====================================================================
// Elementwise helpers
// =====================================================================
__global__ void hgt_f32_to_f16(const float* __restrict__ s, f16* __restrict__ d, long long n)
{
    long long i  = (long long)blockIdx.x * blockDim.x + threadIdx.x;
    long long st = (long long)gridDim.x * blockDim.x;
    for (; i < n; i += st) d[i] = (f16)s[i];
}

// dst[b][n][k] = (f16) src[b][k][n]  — pre-transpose weights for the GEMM B path
__global__ void hgt_f32_to_f16_T(const float* __restrict__ s, f16* __restrict__ d,
                                 int K, int N, long long total)
{
    long long i  = (long long)blockIdx.x * blockDim.x + threadIdx.x;
    long long st = (long long)gridDim.x * blockDim.x;
    const long long kn = (long long)K * N;
    for (; i < total; i += st) {
        long long b = i / kn;
        long long r = i - b * kn;
        int k = (int)(r / N);
        int n = (int)(r - (long long)k * N);
        d[b * kn + (long long)n * K + k] = (f16)s[i];
    }
}

__global__ void hgt_fill_f32(float* __restrict__ p, float v, long long n)
{
    long long i  = (long long)blockIdx.x * blockDim.x + threadIdx.x;
    long long st = (long long)gridDim.x * blockDim.x;
    for (; i < n; i += st) p[i] = v;
}

// =====================================================================
// Edge pass 1: alpha[e,h] = <q[dst,h,:], kr[src,h,:]> * p_rel[h] * scale
// + float atomic-max per (dst,h). 128 threads = 4 waves, wave per head.
// =====================================================================
__global__ __launch_bounds__(128) void hgt_edge_alpha_max(
    const int* __restrict__ ei, int E,
    const float* __restrict__ q, const float* __restrict__ kr,
    const float* __restrict__ prel, float scale,
    float* __restrict__ alpha, float* __restrict__ mx)
{
    int e = blockIdx.x;
    if (e >= E) return;
    int h = threadIdx.x >> 5, lane = threadIdx.x & 31;
    int src = ei[e], dst = ei[E + e];
    const float4* qv = (const float4*)(q  + (size_t)dst * 512 + h * 128);
    const float4* kv = (const float4*)(kr + (size_t)src * 512 + h * 128);
    float4 a = qv[lane], b = kv[lane];
    float p = a.x * b.x + a.y * b.y + a.z * b.z + a.w * b.w;
    #pragma unroll
    for (int s = 16; s > 0; s >>= 1) p += __shfl_xor(p, s);
    if (lane == 0) {
        float al = p * prel[h] * scale;
        alpha[(size_t)e * 4 + h] = al;
        float* addr = mx + (size_t)dst * 4 + h;
        if (al >= 0.0f) atomicMax((int*)addr, __float_as_int(al));
        else            atomicMin((unsigned*)addr, __float_as_uint(al));
    }
}

// =====================================================================
// Edge pass 2: e = exp(alpha - max); s[dst,h] += e; acc[dst,h,:] += e*vr[src,h,:]
// =====================================================================
__global__ __launch_bounds__(128) void hgt_edge_accum(
    const int* __restrict__ ei, int E,
    const float* __restrict__ alpha, const float* __restrict__ mx,
    const float* __restrict__ vr,
    float* __restrict__ ssum, float* __restrict__ acc)
{
    int e = blockIdx.x;
    if (e >= E) return;
    int h = threadIdx.x >> 5, lane = threadIdx.x & 31;
    int src = ei[e], dst = ei[E + e];
    float m = mx[(size_t)dst * 4 + h];
    if (!(m >= -3.0e38f && m <= 3.0e38f)) m = 0.0f;  // mirror where(isfinite(mx),mx,0)
    float ex = expf(alpha[(size_t)e * 4 + h] - m);
    if (lane == 0) atomicAdd(ssum + (size_t)dst * 4 + h, ex);
    const float4* vv = (const float4*)(vr + (size_t)src * 512 + h * 128);
    float4 v = vv[lane];
    float* a = acc + (size_t)dst * 512 + h * 128 + lane * 4;
    atomicAdd(a + 0, ex * v.x);
    atomicAdd(a + 1, ex * v.y);
    atomicAdd(a + 2, ex * v.z);
    atomicAdd(a + 3, ex * v.w);
}

// =====================================================================
// Node: o = gelu(acc / (s + 1e-16)) -> f16 (feeds the Wa GEMM)
// =====================================================================
__global__ void hgt_gelu_div(const float* __restrict__ acc, const float* __restrict__ ssum,
                             f16* __restrict__ out, long long total)
{
    long long i  = (long long)blockIdx.x * blockDim.x + threadIdx.x;
    long long st = (long long)gridDim.x * blockDim.x;
    for (; i < total; i += st) {
        long long n = i >> 9;
        int c = (int)(i & 511);
        int h = c >> 7;
        float x = acc[i] / (ssum[n * 4 + h] + 1e-16f);
        out[i] = (f16)(0.5f * x * (1.0f + erff(x * 0.70710678118654752f)));
    }
}

// =====================================================================
// Node: sigmoid-gated skip + residual + LayerNorm(512) + ReLU.
// One 128-thread block per row; writes fp32 h (in place) + f16 copy.
// =====================================================================
__global__ __launch_bounds__(128) void hgt_mix_ln_relu(
    const float* __restrict__ t, float* __restrict__ h, f16* __restrict__ h16,
    const float* __restrict__ skip, const float* __restrict__ g, const float* __restrict__ bt)
{
    __shared__ float red[128];
    const long long n = blockIdx.x;
    const int tid = threadIdx.x;
    const float sg = 1.0f / (1.0f + expf(-skip[0]));
    float x[4];
    float sum = 0.0f;
    #pragma unroll
    for (int i = 0; i < 4; ++i) {
        int c = tid + i * 128;
        float hv = h[n * 512 + c];
        float m  = sg * t[n * 512 + c] + (1.0f - sg) * hv;  // HGT skip gate
        x[i] = m + hv;                                      // encoder residual
        sum += x[i];
    }
    red[tid] = sum; __syncthreads();
    for (int s = 64; s > 0; s >>= 1) { if (tid < s) red[tid] += red[tid + s]; __syncthreads(); }
    const float mu = red[0] * (1.0f / 512.0f);
    __syncthreads();
    float vs = 0.0f;
    #pragma unroll
    for (int i = 0; i < 4; ++i) { float d = x[i] - mu; vs += d * d; }
    red[tid] = vs; __syncthreads();
    for (int s = 64; s > 0; s >>= 1) { if (tid < s) red[tid] += red[tid + s]; __syncthreads(); }
    const float inv = rsqrtf(red[0] * (1.0f / 512.0f) + 1e-5f);
    #pragma unroll
    for (int i = 0; i < 4; ++i) {
        int c = tid + i * 128;
        float y = (x[i] - mu) * inv * g[c] + bt[c];
        y = fmaxf(y, 0.0f);
        h[n * 512 + c]   = y;
        h16[n * 512 + c] = (f16)y;
    }
}

// =====================================================================
// Host orchestration
// =====================================================================
extern "C" void kernel_launch(void* const* d_in, const int* in_sizes, int n_in,
                              void* d_out, int out_size, void* d_ws, size_t ws_size,
                              hipStream_t stream)
{
    const int IN = 768, HD = 512;
    const int Np = in_sizes[0] / IN;
    const int Na = in_sizes[1] / IN;
    const int Ec = in_sizes[24] / 2;
    const int Ew = in_sizes[25] / 2;
    const int Eb = in_sizes[26] / 2;
    const float scale = 0.08838834764831845f;  // 1/sqrt(128)

    // ---- workspace bump allocator (256B aligned) ----
    char* ws = (char*)d_ws;
    size_t off = 0;
    auto alloc = [&](size_t bytes) -> char* {
        char* p = ws + off;
        off += (bytes + 255) & ~(size_t)255;
        return p;
    };

    f16* xp16  = (f16*)alloc((size_t)Np * IN * 2);
    f16* xa16  = (f16*)alloc((size_t)Na * IN * 2);
    f16* wppT  = (f16*)alloc((size_t)IN * HD * 2);          // [512][768]
    f16* wpaT  = (f16*)alloc((size_t)IN * HD * 2);
    f16* wkT   = (f16*)alloc((size_t)4 * HD * HD * 2);      // 4x [512][512]
    f16* wqT   = (f16*)alloc((size_t)4 * HD * HD * 2);
    f16* wvT   = (f16*)alloc((size_t)4 * HD * HD * 2);
    f16* waT   = (f16*)alloc((size_t)4 * HD * HD * 2);
    f16* arT   = (f16*)alloc((size_t)24 * 128 * 128 * 2);   // 24x [128][128]
    f16* mrT   = (f16*)alloc((size_t)24 * 128 * 128 * 2);
    f16* w256T = (f16*)alloc((size_t)HD * 256 * 2);         // [256][512]
    f16* w128T = (f16*)alloc((size_t)256 * 128 * 2);        // [128][256]

    float* hp   = (float*)alloc((size_t)Np * HD * 4);
    float* ha   = (float*)alloc((size_t)Na * HD * 4);
    f16*   hp16 = (f16*)alloc((size_t)Np * HD * 2);
    f16*   ha16 = (f16*)alloc((size_t)Na * HD * 2);
    f16*   kp16 = (f16*)alloc((size_t)Np * HD * 2);
    f16*   vp16 = (f16*)alloc((size_t)Np * HD * 2);
    f16*   ka16 = (f16*)alloc((size_t)Na * HD * 2);
    f16*   va16 = (f16*)alloc((size_t)Na * HD * 2);
    float* qp   = (float*)alloc((size_t)Np * HD * 4);
    float* qa   = (float*)alloc((size_t)Na * HD * 4);
    float* krP  = (float*)alloc((size_t)Np * HD * 4);  // reused rel0 then rel2
    float* vrP  = (float*)alloc((size_t)Np * HD * 4);
    float* krB  = (float*)alloc((size_t)Na * HD * 4);  // rel1 (author sources)
    float* vrB  = (float*)alloc((size_t)Na * HD * 4);
    float* alC  = (float*)alloc((size_t)Ec * 4 * 4);
    float* alW  = (float*)alloc((size_t)Ew * 4 * 4);
    float* alB  = (float*)alloc((size_t)Eb * 4 * 4);
    float* mxP  = (float*)alloc((size_t)Np * 4 * 4);
    float* sP   = (float*)alloc((size_t)Np * 4 * 4);
    float* mxA  = (float*)alloc((size_t)Na * 4 * 4);
    float* sA   = (float*)alloc((size_t)Na * 4 * 4);
    float* accP = (float*)alloc((size_t)Np * HD * 4);
    float* accA = (float*)alloc((size_t)Na * HD * 4);
    f16*   gp16 = (f16*)alloc((size_t)Np * HD * 2);
    f16*   ga16 = (f16*)alloc((size_t)Na * HD * 2);
    float* tP   = (float*)alloc((size_t)Np * HD * 4);
    float* tA   = (float*)alloc((size_t)Na * HD * 4);
    f16*   h2p  = (f16*)alloc((size_t)Np * 256 * 2);
    f16*   h2a  = (f16*)alloc((size_t)Na * 256 * 2);
    (void)ws_size;

    auto conv = [&](const void* s, f16* d, long long n) {
        int blocks = (int)((n + 255) / 256);
        if (blocks > 16384) blocks = 16384;
        hgt_f32_to_f16<<<blocks, 256, 0, stream>>>((const float*)s, d, n);
    };
    auto convT = [&](const void* s, f16* d, int B, int K, int N) {
        long long n = (long long)B * K * N;
        int blocks = (int)((n + 255) / 256);
        if (blocks > 16384) blocks = 16384;
        hgt_f32_to_f16_T<<<blocks, 256, 0, stream>>>((const float*)s, d, K, N, n);
    };
    auto fill = [&](float* p, float v, long long n) {
        int blocks = (int)((n + 255) / 256);
        if (blocks > 16384) blocks = 16384;
        hgt_fill_f32<<<blocks, 256, 0, stream>>>(p, v, n);
    };
    auto gemm = [&](const f16* A, int lda, long long sAb,
                    const f16* B, int ldb, long long sBb,
                    float* Cf, f16* Ch, int ldc, long long sCb,
                    const float* bias, int M, int N, int K, int relu, int batch) {
        dim3 g((N + 127) / 128, (M + 127) / 128, batch);
        hgt_gemm_f16<<<g, 256, 0, stream>>>(A, lda, sAb, B, ldb, sBb,
                                            Cf, Ch, ldc, sCb, bias, M, N, K, relu);
    };

    // ---- one-time conversions (A-side plain, B-side transposed) ----
    conv (d_in[0],  xp16, (long long)Np * IN);
    conv (d_in[1],  xa16, (long long)Na * IN);
    convT(d_in[2],  wppT,  1, IN, HD);
    convT(d_in[4],  wpaT,  1, IN, HD);
    convT(d_in[6],  wkT,   4, HD, HD);
    convT(d_in[8],  wqT,   4, HD, HD);
    convT(d_in[10], wvT,   4, HD, HD);
    convT(d_in[12], waT,   4, HD, HD);
    convT(d_in[15], arT,  24, 128, 128);
    convT(d_in[16], mrT,  24, 128, 128);
    convT(d_in[20], w256T, 1, HD, 256);
    convT(d_in[22], w128T, 1, 256, 128);

    // ---- input projections: h = x @ Wp + bp (fp32 + f16 copies) ----
    gemm(xp16, IN, 0, wppT, IN, 0, hp, hp16, HD, 0, (const float*)d_in[3], Np, HD, IN, 0, 1);
    gemm(xa16, IN, 0, wpaT, IN, 0, ha, ha16, HD, 0, (const float*)d_in[5], Na, HD, IN, 0, 1);

    const float* bk   = (const float*)d_in[7];
    const float* bq   = (const float*)d_in[9];
    const float* bv   = (const float*)d_in[11];
    const float* ba   = (const float*)d_in[13];
    const float* skip = (const float*)d_in[14];
    const float* prel = (const float*)d_in[17];
    const float* lng  = (const float*)d_in[18];
    const float* lnb  = (const float*)d_in[19];

    for (int l = 0; l < 2; ++l) {
        const size_t wo0 = (size_t)(l * 2 + 0) * HD * HD;
        const size_t wo1 = (size_t)(l * 2 + 1) * HD * HD;

        // K,Q,V per node type (Q kept fp32; K,V as f16 for relation GEMMs)
        gemm(hp16, HD, 0, wkT + wo0, HD, 0, nullptr, kp16, HD, 0, bk + (l*2+0)*HD, Np, HD, HD, 0, 1);
        gemm(hp16, HD, 0, wqT + wo0, HD, 0, qp, nullptr, HD, 0, bq + (l*2+0)*HD, Np, HD, HD, 0, 1);
        gemm(hp16, HD, 0, wvT + wo0, HD, 0, nullptr, vp16, HD, 0, bv + (l*2+0)*HD, Np, HD, HD, 0, 1);
        gemm(ha16, HD, 0, wkT + wo1, HD, 0, nullptr, ka16, HD, 0, bk + (l*2+1)*HD, Na, HD, HD, 0, 1);
        gemm(ha16, HD, 0, wqT + wo1, HD, 0, qa, nullptr, HD, 0, bq + (l*2+1)*HD, Na, HD, HD, 0, 1);
        gemm(ha16, HD, 0, wvT + wo1, HD, 0, nullptr, va16, HD, 0, bv + (l*2+1)*HD, Na, HD, HD, 0, 1);

        // init softmax state
        fill(mxP, -__builtin_inff(), (long long)Np * 4);
        fill(sP, 0.0f, (long long)Np * 4);
        fill(accP, 0.0f, (long long)Np * HD);
        fill(mxA, -__builtin_inff(), (long long)Na * 4);
        fill(sA, 0.0f, (long long)Na * 4);
        fill(accA, 0.0f, (long long)Na * HD);

        // relation transforms, batched per head (A col-block stride 128 in 512 pitch)
        const f16* ar0 = arT + (size_t)(l * 3 + 0) * 4 * 128 * 128;
        const f16* mr0 = mrT + (size_t)(l * 3 + 0) * 4 * 128 * 128;
        const f16* ar1 = arT + (size_t)(l * 3 + 1) * 4 * 128 * 128;
        const f16* mr1 = mrT + (size_t)(l * 3 + 1) * 4 * 128 * 128;
        gemm(kp16, HD, 128, ar0, 128, 16384, krP, nullptr, HD, 128, nullptr, Np, 128, 128, 0, 4);
        gemm(vp16, HD, 128, mr0, 128, 16384, vrP, nullptr, HD, 128, nullptr, Np, 128, 128, 0, 4);
        gemm(ka16, HD, 128, ar1, 128, 16384, krB, nullptr, HD, 128, nullptr, Na, 128, 128, 0, 4);
        gemm(va16, HD, 128, mr1, 128, 16384, vrB, nullptr, HD, 128, nullptr, Na, 128, 128, 0, 4);

        const int* eic = (const int*)d_in[24];
        const int* eiw = (const int*)d_in[25];
        const int* eib = (const int*)d_in[26];

        // paper destinations: segment max over concat(cites, writes), then accumulate
        hgt_edge_alpha_max<<<Ec, 128, 0, stream>>>(eic, Ec, qp, krP, prel + (l*3+0)*4, scale, alC, mxP);
        hgt_edge_alpha_max<<<Ew, 128, 0, stream>>>(eiw, Ew, qp, krB, prel + (l*3+1)*4, scale, alW, mxP);
        hgt_edge_accum<<<Ec, 128, 0, stream>>>(eic, Ec, alC, mxP, vrP, sP, accP);
        hgt_edge_accum<<<Ew, 128, 0, stream>>>(eiw, Ew, alW, mxP, vrB, sP, accP);

        // rel2 (paper->author): reuse krP/vrP buffers
        const f16* ar2 = arT + (size_t)(l * 3 + 2) * 4 * 128 * 128;
        const f16* mr2 = mrT + (size_t)(l * 3 + 2) * 4 * 128 * 128;
        gemm(kp16, HD, 128, ar2, 128, 16384, krP, nullptr, HD, 128, nullptr, Np, 128, 128, 0, 4);
        gemm(vp16, HD, 128, mr2, 128, 16384, vrP, nullptr, HD, 128, nullptr, Np, 128, 128, 0, 4);
        hgt_edge_alpha_max<<<Eb, 128, 0, stream>>>(eib, Eb, qa, krP, prel + (l*3+2)*4, scale, alB, mxA);
        hgt_edge_accum<<<Eb, 128, 0, stream>>>(eib, Eb, alB, mxA, vrP, sA, accA);

        // gelu(acc / (s + 1e-16)) -> f16
        {
            long long nP = (long long)Np * HD, nA = (long long)Na * HD;
            int bP = (int)((nP + 255) / 256); if (bP > 16384) bP = 16384;
            int bA = (int)((nA + 255) / 256); if (bA > 16384) bA = 16384;
            hgt_gelu_div<<<bP, 256, 0, stream>>>(accP, sP, gp16, nP);
            hgt_gelu_div<<<bA, 256, 0, stream>>>(accA, sA, ga16, nA);
        }

        // per-dst-type output linear
        gemm(gp16, HD, 0, waT + wo0, HD, 0, tP, nullptr, HD, 0, ba + (l*2+0)*HD, Np, HD, HD, 0, 1);
        gemm(ga16, HD, 0, waT + wo1, HD, 0, tA, nullptr, HD, 0, ba + (l*2+1)*HD, Na, HD, HD, 0, 1);

        // gated skip + residual + LayerNorm + ReLU -> new h (fp32 + f16)
        hgt_mix_ln_relu<<<Np, 128, 0, stream>>>(tP, hp, hp16, skip + l*2+0, lng + (size_t)l*HD, lnb + (size_t)l*HD);
        hgt_mix_ln_relu<<<Na, 128, 0, stream>>>(tA, ha, ha16, skip + l*2+1, lng + (size_t)l*HD, lnb + (size_t)l*HD);
    }

    // ---- final MLP: relu(h @ W256 + b256) @ W128 + b128 -> d_out ----
    const float* b256 = (const float*)d_in[21];
    const float* b128 = (const float*)d_in[23];
    gemm(hp16, HD, 0, w256T, HD, 0, nullptr, h2p, 256, 0, b256, Np, 256, HD, 1, 1);
    gemm(ha16, HD, 0, w256T, HD, 0, nullptr, h2a, 256, 0, b256, Na, 256, HD, 1, 1);
    float* outp = (float*)d_out;
    gemm(h2p, 256, 0, w128T, 256, 0, outp, nullptr, 128, 0, b128, Np, 128, 256, 0, 1);
    gemm(h2a, 256, 0, w128T, 256, 0, outp + (size_t)Np * 128, nullptr, 128, 0, b128, Na, 128, 256, 0, 1);
}
